// MoE_72971494359533
// MI455X (gfx1250) — compile-verified, matrix operand order checked by reading, not run on hardware
//
#include <hip/hip_runtime.h>
#include <stdint.h>

// ---------------- problem constants (match reference) ----------------
#define NE    8       // experts
#define CD    1024    // embed dim
#define FED   1024    // expert inter dim
#define FSD   4096    // shared inter dim
#define NTOK  8192    // B*T = 4*2048

// ---------------- GEMM tiling ----------------
#define BM 128
#define BN 64
#define BK 32

// Tensor Data Mover staging via inline asm (portable across toolchains; the
// clang builtin has different arity on ROCm 7.2 vs upstream). Set to 0 to fall
// back to vector-load staging.
#define USE_TDM 1

typedef __attribute__((ext_vector_type(16))) __bf16          v16bf;
typedef __attribute__((ext_vector_type(16))) unsigned short  v16us;
typedef __attribute__((ext_vector_type(8)))  unsigned short  v8us;
typedef __attribute__((ext_vector_type(8)))  float           v8f;
typedef __attribute__((ext_vector_type(4)))  unsigned int    v4u;
typedef __attribute__((ext_vector_type(8)))  unsigned int    v8u;

// round-to-nearest-even f32 -> bf16 (bit pattern)
__device__ __forceinline__ unsigned short f2bf(float f) {
  unsigned int u = __float_as_uint(f);
  unsigned int r = u + 0x7fffu + ((u >> 16) & 1u);
  return (unsigned short)(r >> 16);
}

// Build a 16x32 bf16 WMMA fragment from an LDS tile whose rows are BK=32
// bf16 elements. Per ISA 7.12.2: lane<16 holds K{0..7,16..23} of row M=lane,
// lane>=16 holds K{8..15,24..31}. Both halves are contiguous 16B chunks.
__device__ __forceinline__ v16bf load_frag(const unsigned short* sm, int row, int hh) {
  v8us lo = *(const v8us*)(sm + row * BK + hh * 8);
  v8us hi = *(const v8us*)(sm + row * BK + 16 + hh * 8);
  v16us t;
#pragma unroll
  for (int i = 0; i < 8; ++i) { t[i] = lo[i]; t[i + 8] = hi[i]; }
  return __builtin_bit_cast(v16bf, t);
}

__device__ __forceinline__ v8f wmma_bf16(v16bf a, v16bf b, v8f c) {
  // (neg_a, A, neg_b, B, c_mod, C, reuse_a, reuse_b)
  return __builtin_amdgcn_wmma_f32_16x16x32_bf16(false, a, false, b, (short)0, c,
                                                 false, false);
}

#if USE_TDM
// Wait until TENSORcnt <= n (immediate).
#define WAIT_TENSORCNT(n) asm volatile("s_wait_tensorcnt " #n ::: "memory")

// Defeat LLVM's "LDS object is never stored -> loads are undef" folding (which
// deleted the ds_loads in round 3; volatile loads instead lowered to serialized
// SYS-scope flat loads in round 4). One store to an opaque index (asm output,
// actually 0) may alias any element, so loads from sMem must stay. The store
// lands before the first TDM issue and is overwritten by the first tile.
#define LDS_GUARD(base, tid)                                     \
  do {                                                           \
    unsigned _gidx;                                              \
    asm("s_mov_b32 %0, 0" : "=s"(_gidx));                        \
    if ((tid) == 0) (base)[_gidx] = (unsigned short)0;           \
    __syncthreads();                                             \
  } while (0)

// Issue a TDM 2D-tile load: tile of (rows x BK) bf16 elements from a row-major
// matrix with leading dimension ld_elems, starting at gptr, into LDS byte
// offset lds_byte_off. D# layout per CDNA5 ISA ch.8 (groups 2/3 zero: 2D tile).
// All descriptor words are wave-uniform -> SGPR groups via the "s" constraint.
__device__ __forceinline__ void tdm_load_tile(unsigned lds_byte_off,
                                              const unsigned short* gptr,
                                              int rows, int ld_elems) {
  unsigned long long ga = (unsigned long long)(uintptr_t)gptr;
  v4u g0;
  g0[0] = 1u;                                        // count=1, user descriptor
  g0[1] = lds_byte_off;                              // lds_addr
  g0[2] = (unsigned)ga;                              // global_addr[31:0]
  g0[3] = ((unsigned)(ga >> 32) & 0x01FFFFFFu)       // global_addr[56:32]
          | (2u << 30);                              // type=2 ("image")
  v8u g1;
  g1[0] = 1u << 16;                                  // data_size=2B (bf16)
  g1[1] = (unsigned)BK << 16;                        // tensor_dim0 = BK
  g1[2] = (unsigned)rows << 16;                      // tensor_dim1 = rows
  g1[3] = (unsigned)BK << 16;                        // tile_dim0 = BK
  g1[4] = (unsigned)rows;                            // tile_dim1 = rows
  g1[5] = (unsigned)ld_elems;                        // tensor_dim0_stride = ld
  g1[6] = 0u;
  g1[7] = 0u;
  v4u gz = {0u, 0u, 0u, 0u};
  asm volatile("tensor_load_to_lds %0, %1, %2, %3"
               :
               : "s"(g0), "s"(g1), "s"(gz), "s"(gz)
               : "memory");
}
#endif

// ---------------- elementwise cast x (f32 -> bf16), float4 wide ----------------
__global__ void cast_x_kernel(const float* __restrict__ in,
                              unsigned short* __restrict__ out, int n4) {
  int i = blockIdx.x * blockDim.x + threadIdx.x;
  if (i >= n4) return;
  float4 v = ((const float4*)in)[i];
  unsigned lo = (unsigned)f2bf(v.x) | ((unsigned)f2bf(v.y) << 16);
  unsigned hi = (unsigned)f2bf(v.z) | ((unsigned)f2bf(v.w) << 16);
  ((uint2*)out)[i] = make_uint2(lo, hi);
}

// ---------------- transpose + cast: in f32 [R,Cc] -> out bf16 [Cc,R], batched z --------
__global__ void transpose_cast_kernel(const float* __restrict__ in,
                                      unsigned short* __restrict__ out,
                                      int R, int Cc) {
  __shared__ float t[32][33];
  size_t bo = (size_t)blockIdx.z * R * Cc;
  in += bo; out += bo;
  int tx = threadIdx.x, ty = threadIdx.y;
  int r0 = blockIdx.y * 32, c0 = blockIdx.x * 32;
#pragma unroll
  for (int i = 0; i < 4; ++i)
    t[ty + 8 * i][tx] = in[(size_t)(r0 + ty + 8 * i) * Cc + c0 + tx];
  __syncthreads();
#pragma unroll
  for (int i = 0; i < 4; ++i)
    out[(size_t)(c0 + ty + 8 * i) * R + r0 + tx] = f2bf(t[tx][ty + 8 * i]);
}

// ---------------- gating: one wave32 per token ----------------
__global__ void gate_kernel(const float* __restrict__ x, const float* __restrict__ gw,
                            float* __restrict__ combine, float* __restrict__ scores,
                            int ntok) {
  int gt = blockIdx.x * blockDim.x + threadIdx.x;
  int tok = gt >> 5, lane = gt & 31;
  if (tok >= ntok) return;
  float a[NE];
#pragma unroll
  for (int e = 0; e < NE; ++e) a[e] = 0.f;
  const float* xr = x + (size_t)tok * CD;
  for (int c = lane; c < CD; c += 32) {
    float xv = xr[c];
#pragma unroll
    for (int e = 0; e < NE; ++e) a[e] += xv * gw[e * CD + c];
  }
#pragma unroll
  for (int e = 0; e < NE; ++e)
    for (int off = 16; off > 0; off >>= 1) a[e] += __shfl_xor(a[e], off, 32);
  if (lane == 0) {
    float m = a[0];
#pragma unroll
    for (int e = 1; e < NE; ++e) m = fmaxf(m, a[e]);
    float p[NE], s = 0.f;
#pragma unroll
    for (int e = 0; e < NE; ++e) { p[e] = __expf(a[e] - m); s += p[e]; }
    float inv = 1.f / s;
#pragma unroll
    for (int e = 0; e < NE; ++e) p[e] *= inv;
    int i1 = 0;
#pragma unroll
    for (int e = 1; e < NE; ++e) if (p[e] > p[i1]) i1 = e;
    int i2 = (i1 == 0) ? 1 : 0;
#pragma unroll
    for (int e = 0; e < NE; ++e) if (e != i1 && p[e] > p[i2]) i2 = e;
#pragma unroll
    for (int e = 0; e < NE; ++e) {
      combine[(size_t)tok * NE + e] = (e == i1) ? p[i1] : ((e == i2) ? p[i2] : 0.f);
      scores[(size_t)tok * NE + e] = a[e];
    }
  }
}

// ---------------- GEMM1: H = silu(A@W1) * (A@W2), bf16 in/out ----------------
// A [M,K] lda ; B1,B2 transposed: rows = output-F, length K (ldb) ; H [M,F] ldh
__global__ __launch_bounds__(256) void gemm_swiglu_kernel(
    const unsigned short* __restrict__ A, const unsigned short* __restrict__ B1,
    const unsigned short* __restrict__ B2, unsigned short* __restrict__ H,
    int Kd, int lda, int ldb, int ldh) {
#define STAGE1 ((BM + 2 * BN) * BK)  // elements per buffer
  __shared__ __align__(128) unsigned short sMem[2 * STAGE1];
  const int tid = threadIdx.x;
  const int m0 = blockIdx.y * BM, n0 = blockIdx.x * BN;
  const int wave = tid >> 5, lane = tid & 31;
  const int wrow = wave & 3, wcol = wave >> 2;  // 4x2 wave grid, 32x32 per wave
  const int lr = lane & 15, hh = lane >> 4;

  v8f acc1[2][2], acc2[2][2];
#pragma unroll
  for (int mt = 0; mt < 2; ++mt)
#pragma unroll
    for (int nt = 0; nt < 2; ++nt) {
      acc1[mt][nt] = (v8f){0.f, 0.f, 0.f, 0.f, 0.f, 0.f, 0.f, 0.f};
      acc2[mt][nt] = (v8f){0.f, 0.f, 0.f, 0.f, 0.f, 0.f, 0.f, 0.f};
    }

  const int nsteps = Kd / BK;
#if USE_TDM
  // sMem is the only LDS object -> it sits at LDS offset 0; buffer b offsets:
  //   A  tile: b*STAGE1*2
  //   B1 tile: b*STAGE1*2 + BM*BK*2
  //   B2 tile: b*STAGE1*2 + (BM+BN)*BK*2
  LDS_GUARD(sMem, tid);
  if (wave == 0) {
    tdm_load_tile(0u, A + (size_t)m0 * lda, BM, lda);
    tdm_load_tile(BM * BK * 2u, B1 + (size_t)n0 * ldb, BN, ldb);
    tdm_load_tile((BM + BN) * BK * 2u, B2 + (size_t)n0 * ldb, BN, ldb);
  }
  for (int s = 0; s < nsteps; ++s) {
    const int b = s & 1;
    const unsigned short* sA  = sMem + b * STAGE1;
    const unsigned short* sB1 = sA + BM * BK;
    const unsigned short* sB2 = sA + (BM + BN) * BK;
    if (wave == 0) {
      if (s + 1 < nsteps) {
        const int kt = (s + 1) * BK;
        const unsigned bb = (unsigned)((b ^ 1) * STAGE1 * 2);
        tdm_load_tile(bb, A + (size_t)m0 * lda + kt, BM, lda);
        tdm_load_tile(bb + BM * BK * 2u, B1 + (size_t)n0 * ldb + kt, BN, ldb);
        tdm_load_tile(bb + (BM + BN) * BK * 2u, B2 + (size_t)n0 * ldb + kt, BN, ldb);
        WAIT_TENSORCNT(0x3);  // only the older 3 transfers must be done
      } else {
        WAIT_TENSORCNT(0x0);
      }
    }
    __syncthreads();
#else
  for (int s = 0; s < nsteps; ++s) {
    const int kt = s * BK;
    unsigned short* sA  = sMem;
    unsigned short* sB1 = sMem + BM * BK;
    unsigned short* sB2 = sMem + (BM + BN) * BK;
#pragma unroll
    for (int i = 0; i < 2; ++i) {
      int c = tid + i * 256;
      int row = c >> 2, cc = (c & 3) * 8;
      *(uint4*)(sA + row * BK + cc) =
          *(const uint4*)(A + (size_t)(m0 + row) * lda + kt + cc);
    }
    {
      int row = tid >> 2, cc = (tid & 3) * 8;
      size_t gb = (size_t)(n0 + row) * ldb + kt + cc;
      *(uint4*)(sB1 + row * BK + cc) = *(const uint4*)(B1 + gb);
      *(uint4*)(sB2 + row * BK + cc) = *(const uint4*)(B2 + gb);
    }
    if (kt + BK < Kd) {  // lowers to global_prefetch_b8
      __builtin_prefetch(A + (size_t)(m0 + (tid >> 1)) * lda + kt + BK, 0, 3);
      __builtin_prefetch(B1 + (size_t)(n0 + (tid >> 2)) * ldb + kt + BK, 0, 3);
      __builtin_prefetch(B2 + (size_t)(n0 + (tid >> 2)) * ldb + kt + BK, 0, 3);
    }
    __syncthreads();
#endif
    v16bf af[2], b1f[2], b2f[2];
#pragma unroll
    for (int mt = 0; mt < 2; ++mt) af[mt] = load_frag(sA, wrow * 32 + mt * 16 + lr, hh);
#pragma unroll
    for (int nt = 0; nt < 2; ++nt) {
      b1f[nt] = load_frag(sB1, wcol * 32 + nt * 16 + lr, hh);
      b2f[nt] = load_frag(sB2, wcol * 32 + nt * 16 + lr, hh);
    }
#pragma unroll
    for (int mt = 0; mt < 2; ++mt)
#pragma unroll
      for (int nt = 0; nt < 2; ++nt) {
        acc1[mt][nt] = wmma_bf16(af[mt], b1f[nt], acc1[mt][nt]);
        acc2[mt][nt] = wmma_bf16(af[mt], b2f[nt], acc2[mt][nt]);
      }
    __syncthreads();
  }

  // epilogue: SwiGLU, store bf16
#pragma unroll
  for (int mt = 0; mt < 2; ++mt)
#pragma unroll
    for (int nt = 0; nt < 2; ++nt) {
      int col = n0 + wcol * 32 + nt * 16 + lr;
#pragma unroll
      for (int i = 0; i < 8; ++i) {
        int row = m0 + wrow * 32 + mt * 16 + hh * 8 + i;
        float s1 = acc1[mt][nt][i], s2 = acc2[mt][nt][i];
        float h = (s1 / (1.f + __expf(-s1))) * s2;
        H[(size_t)row * ldh + col] = f2bf(h);
      }
    }
#undef STAGE1
}

// ---------------- GEMM2: Y (+)= (A@Wp) * combine[row] ----------------
// A [M,K] bf16 lda ; B transposed: rows = output-C, length K along ldb ; Y f32 [M,ldy]
__global__ __launch_bounds__(256) void gemm_proj_kernel(
    const unsigned short* __restrict__ A, const unsigned short* __restrict__ B,
    float* __restrict__ Y, const float* __restrict__ combine, int eidx, int accFlag,
    int Kd, int lda, int ldb, int ldy) {
#define STAGE2 ((BM + BN) * BK)
  __shared__ __align__(128) unsigned short sMem[2 * STAGE2];
  const int tid = threadIdx.x;
  const int m0 = blockIdx.y * BM, n0 = blockIdx.x * BN;
  const int wave = tid >> 5, lane = tid & 31;
  const int wrow = wave & 3, wcol = wave >> 2;
  const int lr = lane & 15, hh = lane >> 4;

  v8f acc[2][2];
#pragma unroll
  for (int mt = 0; mt < 2; ++mt)
#pragma unroll
    for (int nt = 0; nt < 2; ++nt)
      acc[mt][nt] = (v8f){0.f, 0.f, 0.f, 0.f, 0.f, 0.f, 0.f, 0.f};

  const int nsteps = Kd / BK;
#if USE_TDM
  LDS_GUARD(sMem, tid);
  if (wave == 0) {
    tdm_load_tile(0u, A + (size_t)m0 * lda, BM, lda);
    tdm_load_tile(BM * BK * 2u, B + (size_t)n0 * ldb, BN, ldb);
  }
  for (int s = 0; s < nsteps; ++s) {
    const int b = s & 1;
    const unsigned short* sA = sMem + b * STAGE2;
    const unsigned short* sB = sA + BM * BK;
    if (wave == 0) {
      if (s + 1 < nsteps) {
        const int kt = (s + 1) * BK;
        const unsigned bb = (unsigned)((b ^ 1) * STAGE2 * 2);
        tdm_load_tile(bb, A + (size_t)m0 * lda + kt, BM, lda);
        tdm_load_tile(bb + BM * BK * 2u, B + (size_t)n0 * ldb + kt, BN, ldb);
        WAIT_TENSORCNT(0x2);
      } else {
        WAIT_TENSORCNT(0x0);
      }
    }
    __syncthreads();
#else
  for (int s = 0; s < nsteps; ++s) {
    const int kt = s * BK;
    unsigned short* sA = sMem;
    unsigned short* sB = sMem + BM * BK;
#pragma unroll
    for (int i = 0; i < 2; ++i) {
      int c = tid + i * 256;
      int row = c >> 2, cc = (c & 3) * 8;
      *(uint4*)(sA + row * BK + cc) =
          *(const uint4*)(A + (size_t)(m0 + row) * lda + kt + cc);
    }
    {
      int row = tid >> 2, cc = (tid & 3) * 8;
      *(uint4*)(sB + row * BK + cc) =
          *(const uint4*)(B + (size_t)(n0 + row) * ldb + kt + cc);
    }
    if (kt + BK < Kd) {
      __builtin_prefetch(A + (size_t)(m0 + (tid >> 1)) * lda + kt + BK, 0, 3);
      __builtin_prefetch(B + (size_t)(n0 + (tid >> 2)) * ldb + kt + BK, 0, 3);
    }
    __syncthreads();
#endif
    v16bf af[2], bf[2];
#pragma unroll
    for (int mt = 0; mt < 2; ++mt) af[mt] = load_frag(sA, wrow * 32 + mt * 16 + lr, hh);
#pragma unroll
    for (int nt = 0; nt < 2; ++nt) bf[nt] = load_frag(sB, wcol * 32 + nt * 16 + lr, hh);
#pragma unroll
    for (int mt = 0; mt < 2; ++mt)
#pragma unroll
      for (int nt = 0; nt < 2; ++nt)
        acc[mt][nt] = wmma_bf16(af[mt], bf[nt], acc[mt][nt]);
    __syncthreads();
  }

#pragma unroll
  for (int mt = 0; mt < 2; ++mt)
#pragma unroll
    for (int nt = 0; nt < 2; ++nt) {
      int col = n0 + wcol * 32 + nt * 16 + lr;
#pragma unroll
      for (int i = 0; i < 8; ++i) {
        int row = m0 + wrow * 32 + mt * 16 + hh * 8 + i;
        float v = acc[mt][nt][i];
        if (combine) v *= combine[(size_t)row * NE + eidx];
        float* yp = Y + (size_t)row * ldy + col;
        if (accFlag) v += *yp;
        *yp = v;
      }
    }
#undef STAGE2
}

// ---------------- host-side orchestration ----------------
extern "C" void kernel_launch(void* const* d_in, const int* in_sizes, int n_in,
                              void* d_out, int out_size, void* d_ws, size_t ws_size,
                              hipStream_t stream) {
  const float* x      = (const float*)d_in[0];  // [N, C]
  const float* gate_w = (const float*)d_in[1];  // [E, C]
  const float* ew1    = (const float*)d_in[2];  // [E, C, FE]
  const float* ew2    = (const float*)d_in[3];  // [E, C, FE]
  const float* ewp    = (const float*)d_in[4];  // [E, FE, C]
  const float* sw1    = (const float*)d_in[5];  // [C, FS]
  const float* sw2    = (const float*)d_in[6];  // [C, FS]
  const float* swp    = (const float*)d_in[7];  // [FS, C]
  float* out = (float*)d_out;                   // [N*C] y+z, then [N*E] scores

  // workspace carve-out (~104 MB total)
  char* w = (char*)d_ws;
  float* combine = (float*)w;            w += (size_t)NTOK * NE * 4;
  unsigned short* xb   = (unsigned short*)w; w += (size_t)NTOK * CD * 2;
  unsigned short* ew1t = (unsigned short*)w; w += (size_t)NE * FED * CD * 2;
  unsigned short* ew2t = (unsigned short*)w; w += (size_t)NE * FED * CD * 2;
  unsigned short* ewpt = (unsigned short*)w; w += (size_t)NE * CD * FED * 2;
  unsigned short* sw1t = (unsigned short*)w; w += (size_t)FSD * CD * 2;
  unsigned short* sw2t = (unsigned short*)w; w += (size_t)FSD * CD * 2;
  unsigned short* swpt = (unsigned short*)w; w += (size_t)CD * FSD * 2;
  unsigned short* hb   = (unsigned short*)w; w += (size_t)NTOK * FED * 2;

  dim3 t32x8(32, 8);

  // 1) precision conversion / weight transposition (bf16, K-contiguous rows)
  cast_x_kernel<<<(NTOK * CD / 4 + 255) / 256, 256, 0, stream>>>(x, xb, NTOK * CD / 4);
  transpose_cast_kernel<<<dim3(FED / 32, CD / 32, NE), t32x8, 0, stream>>>(ew1, ew1t, CD, FED);
  transpose_cast_kernel<<<dim3(FED / 32, CD / 32, NE), t32x8, 0, stream>>>(ew2, ew2t, CD, FED);
  transpose_cast_kernel<<<dim3(CD / 32, FED / 32, NE), t32x8, 0, stream>>>(ewp, ewpt, FED, CD);
  transpose_cast_kernel<<<dim3(FSD / 32, CD / 32, 1), t32x8, 0, stream>>>(sw1, sw1t, CD, FSD);
  transpose_cast_kernel<<<dim3(FSD / 32, CD / 32, 1), t32x8, 0, stream>>>(sw2, sw2t, CD, FSD);
  transpose_cast_kernel<<<dim3(CD / 32, FSD / 32, 1), t32x8, 0, stream>>>(swp, swpt, FSD, CD);

  // 2) gating: combine weights + raw scores (second output)
  gate_kernel<<<NTOK * 32 / 256, 256, 0, stream>>>(x, gate_w, combine,
                                                   out + (size_t)NTOK * CD, NTOK);

  // 3) experts (masked-dense, like the reference): expert 0 writes, rest accumulate
  for (int e = 0; e < NE; ++e) {
    gemm_swiglu_kernel<<<dim3(FED / BN, NTOK / BM), 256, 0, stream>>>(
        xb, ew1t + (size_t)e * FED * CD, ew2t + (size_t)e * FED * CD, hb,
        CD, CD, CD, FED);
    gemm_proj_kernel<<<dim3(CD / BN, NTOK / BM), 256, 0, stream>>>(
        hb, ewpt + (size_t)e * CD * FED, out, combine, e, e ? 1 : 0,
        FED, FED, FED, CD);
  }

  // 4) shared FFN: FS=4096 processed as 4 K-chunks of 1024 reusing hb
  for (int c = 0; c < 4; ++c) {
    gemm_swiglu_kernel<<<dim3(1024 / BN, NTOK / BM), 256, 0, stream>>>(
        xb, sw1t + (size_t)c * 1024 * CD, sw2t + (size_t)c * 1024 * CD, hb,
        CD, CD, CD, 1024);
    gemm_proj_kernel<<<dim3(CD / BN, NTOK / BM), 256, 0, stream>>>(
        hb, swpt + (size_t)c * 1024, out, nullptr, 0, 1,
        1024, 1024, FSD, CD);
  }
}
